// DisentangledSelfAttentionInteractingLayer_87162066305264
// MI455X (gfx1250) — compile-verified, matrix operand order checked by reading, not run on hardware
//
#include <hip/hip_runtime.h>
#include <stdint.h>

// ---------------------------------------------------------------------------
// Disentangled self-attention interacting layer, fused, for MI455X (gfx1250).
//
// BS=4096, F=E=A=128, NH=2, dph=64. One 256-thread (8 x wave32) workgroup per
// batch sample. X and the weights are converted to bf16 in LDS; all GEMMs use
// v_wmma_f32_16x16x32_bf16 (f32 accumulate). Softmax / means / bias adds in
// f32. Residual projection is kept in LDS so global traffic is just
// read-X + write-out (~0.54 GB total -> ~23 us at 23.3 TB/s roofline).
//
// Reshape algebra (torch .view semantics):
//   (f, a) -> h = f>>6, n = 2*(f&63) + (a>>6), d = a&63
//   unary softmax value for (h, n) lives at flat index h*128 + n
// Q/K biases and the unary bias cancel (mean-centering / softmax shift
// invariance), so only the V and residual biases are applied.
// ---------------------------------------------------------------------------

typedef __attribute__((ext_vector_type(16))) __bf16 v16bf;
typedef __attribute__((ext_vector_type(8)))  float  v8f;

namespace {

constexpr int SX_STRIDE = 136;  // halves (128 + 8 pad) for 128-wide bf16 mats
constexpr int MU_STRIDE = 72;   // halves (64 + 8 pad) for muQ/muK
constexpr int R_STRIDE  = 132;  // floats (128 + 4 pad) for residual buffer

constexpr size_t SZ_128   = size_t(128) * SX_STRIDE * 2;  // 34816 B
constexpr size_t SZ_MU    = size_t(256) * MU_STRIDE * 2;  // 36864 B (2 heads)
constexpr size_t SZ_R     = size_t(128) * R_STRIDE * 4;   // 67584 B

constexpr size_t OFF_X     = 0;                  // bf16 [128][136]  (aliased by attn later)
constexpr size_t OFF_W     = OFF_X    + SZ_128;  // bf16 [128][136]  (aliased by attn later)
constexpr size_t OFF_MUQ   = OFF_W    + SZ_128;  // bf16 [2*128][72]
constexpr size_t OFF_MUK   = OFF_MUQ  + SZ_MU;   // bf16 [2*128][72]
constexpr size_t OFF_VT    = OFF_MUK  + SZ_MU;   // bf16 [2*64][136]  V^T per head (+bias)
constexpr size_t OFF_R     = OFF_VT   + SZ_128;  // f32  [128][132]  residual projection
constexpr size_t OFF_COL   = OFF_R    + SZ_R;    // f32  [2][128] column sums
constexpr size_t OFF_MEANQ = OFF_COL  + 1024;    // f32  [2][64]
constexpr size_t OFF_MEANK = OFF_MEANQ + 512;    // f32  [2][64]
constexpr size_t OFF_US    = OFF_MEANK + 512;    // f32  [256] unary scores
constexpr size_t OFF_U     = OFF_US   + 1024;    // f32  [256] unary softmax (flat h*128+n)
constexpr size_t OFF_UW    = OFF_U    + 1024;    // f32  [256] staged unary_w
constexpr size_t OFF_BIAS  = OFF_UW   + 1024;    // f32  [128] staged bias
constexpr size_t SMEM_BYTES = OFF_BIAS + 512;    // 251392 B < 320 KB

__device__ __forceinline__ __bf16 f2bf(float x) {
  union { float f; uint32_t u; } v; v.f = x;
  uint32_t r = v.u + 0x7FFFu + ((v.u >> 16) & 1u);   // round-to-nearest-even
  uint16_t h = uint16_t(r >> 16);
  __bf16 o; __builtin_memcpy(&o, &h, sizeof(o));
  return o;
}
__device__ __forceinline__ float bf2f(__bf16 x) {
  uint16_t h; __builtin_memcpy(&h, &x, sizeof(h));
  union { uint32_t u; float f; } v; v.u = uint32_t(h) << 16;
  return v.f;
}

union FragCast { v16bf v; uint4 q[2]; };

// Load one 16x32 bf16 WMMA operand fragment from a row-major LDS matrix.
// ISA A-layout (16-bit, 16x32): lane L -> row = L%16,
// K elements = base {0..7, 16..23} + 8*(L>=16)  => two 16-byte LDS reads.
// The B operand of D = A x B^T is addressed identically on the row-major
// right-hand matrix (its fragment columns are that matrix's rows).
__device__ __forceinline__ v16bf load_frag(const __bf16* __restrict__ base,
                                           int row0, int k0, int stride, int lane) {
  const int r  = row0 + (lane & 15);
  const int kk = k0 + ((lane & 16) >> 1);        // +8 halves for upper half-wave
  const __bf16* p = base + r * stride + kk;
  FragCast f;
  f.q[0] = *reinterpret_cast<const uint4*>(p);        // K = kk .. kk+7
  f.q[1] = *reinterpret_cast<const uint4*>(p + 16);   // K = kk+16 .. kk+23
  return f.v;
}

__device__ __forceinline__ void zero_acc(v8f* acc, int n) {
  for (int i = 0; i < n; ++i) acc[i] = v8f{0.f, 0.f, 0.f, 0.f, 0.f, 0.f, 0.f, 0.f};
}

// One wave computes a 16-row strip of D = A * B^T (A: [*,K], B: [*,K], both
// row-major bf16 in LDS). NT = number of 16-col tiles, KT = K/32.
template <int NT, int KT>
__device__ __forceinline__ void gemm_strip(const __bf16* __restrict__ A, int sA,
                                           const __bf16* __restrict__ B, int sB,
                                           int mtile, v8f* acc, int lane) {
#pragma unroll
  for (int kt = 0; kt < KT; ++kt) {
    v16bf af = load_frag(A, mtile * 16, kt * 32, sA, lane);
#pragma unroll
    for (int nt = 0; nt < NT; ++nt) {
      v16bf bfr = load_frag(B, nt * 16, kt * 32, sB, lane);
      acc[nt] = __builtin_amdgcn_wmma_f32_16x16x32_bf16(
          false, af, false, bfr, (short)0, acc[nt], false, false);
    }
  }
}

// Stage a 128x128 f32 global matrix into LDS as bf16 (row stride SX_STRIDE).
__device__ __forceinline__ void stage128(__bf16* dst, const float* __restrict__ src,
                                         int tid) {
  const float4* s4 = reinterpret_cast<const float4*>(src);
#pragma unroll 4
  for (int i = tid; i < 128 * 32; i += 256) {
    float4 v = s4[i];
    int row = i >> 5;
    int col = (i & 31) << 2;
    __bf16* d = dst + row * SX_STRIDE + col;
    d[0] = f2bf(v.x); d[1] = f2bf(v.y); d[2] = f2bf(v.z); d[3] = f2bf(v.w);
  }
}

// Q or K projection: GEMM, per-(head,d) column means via shuffle + LDS atomic,
// then mean-centered bf16 store in the per-head reshaped [n][d] layout.
__device__ __forceinline__ void qk_project(const float* __restrict__ gW,
                                           __bf16* sX, __bf16* sW, __bf16* sMu,
                                           float* sMean, float* sCol,
                                           int tid, int lane, int wave) {
  stage128(sW, gW, tid);
  sCol[tid] = 0.f;
  __syncthreads();

  v8f acc[8]; zero_acc(acc, 8);
  gemm_strip<8, 4>(sX, SX_STRIDE, sW, SX_STRIDE, wave, acc, lane);

  const int h = wave >> 2;  // this wave's 16 f-rows all lie in head wave/4
#pragma unroll
  for (int nt = 0; nt < 8; ++nt) {
    float s = 0.f;
#pragma unroll
    for (int r = 0; r < 8; ++r) s += acc[nt][r];
    s += __shfl_xor(s, 16, 32);              // combine row-halves of the tile
    if (lane < 16) atomicAdd(&sCol[h * 128 + nt * 16 + lane], s);
  }
  __syncthreads();

  if (tid < 128) {  // mean over n mixes columns d and d+64 of the head strip
    int hh = tid >> 6, d = tid & 63;
    sMean[tid] = (sCol[hh * 128 + d] + sCol[hh * 128 + 64 + d]) * (1.0f / 128.0f);
  }
  __syncthreads();

#pragma unroll
  for (int nt = 0; nt < 8; ++nt)
#pragma unroll
    for (int r = 0; r < 8; ++r) {
      int a  = nt * 16 + (lane & 15);
      int f  = wave * 16 + r + ((lane & 16) >> 1);
      int hh = f >> 6;
      int n  = ((f & 63) << 1) + (a >> 6);
      int d  = a & 63;
      sMu[(hh * 128 + n) * MU_STRIDE + d] = f2bf(acc[nt][r] - sMean[hh * 64 + d]);
    }
  __syncthreads();
}

}  // namespace

__global__ __launch_bounds__(256) void disattn_kernel(
    const float* __restrict__ gin, const float* __restrict__ g_uw,
    const float* __restrict__ g_wq, const float* __restrict__ g_wk,
    const float* __restrict__ g_wv, const float* __restrict__ g_wvb,
    const float* __restrict__ g_wr, const float* __restrict__ g_wrb,
    float* __restrict__ gout) {
  extern __shared__ __align__(16) char smem[];
  __bf16* sX    = reinterpret_cast<__bf16*>(smem + OFF_X);
  __bf16* sW    = reinterpret_cast<__bf16*>(smem + OFF_W);
  __bf16* sMuQ  = reinterpret_cast<__bf16*>(smem + OFF_MUQ);
  __bf16* sMuK  = reinterpret_cast<__bf16*>(smem + OFF_MUK);
  __bf16* sVt   = reinterpret_cast<__bf16*>(smem + OFF_VT);
  float*  sR    = reinterpret_cast<float*>(smem + OFF_R);
  float*  sCol  = reinterpret_cast<float*>(smem + OFF_COL);
  float*  sMeanQ= reinterpret_cast<float*>(smem + OFF_MEANQ);
  float*  sMeanK= reinterpret_cast<float*>(smem + OFF_MEANK);
  float*  sUS   = reinterpret_cast<float*>(smem + OFF_US);
  float*  sU    = reinterpret_cast<float*>(smem + OFF_U);
  float*  sUW   = reinterpret_cast<float*>(smem + OFF_UW);
  float*  sBias = reinterpret_cast<float*>(smem + OFF_BIAS);
  __bf16* sAttn = reinterpret_cast<__bf16*>(smem + OFF_X);  // aliases X+W regions

  const int tid  = threadIdx.x;
  const int lane = tid & 31;
  const int wave = tid >> 5;
  const size_t base = size_t(blockIdx.x) << 14;  // b * 128 * 128

  // ---- Phase 0: stage X (bf16) and unary weights -------------------------
  stage128(sX, gin + base, tid);
  sUW[tid] = g_uw[tid];
  __syncthreads();

  // ---- Phase 0b: unary scores + softmax over the field dim ---------------
  {
    int f = tid >> 1, c = tid & 1;
    const __bf16* xr = sX + f * SX_STRIDE;
    const float*  w  = sUW + c * 128;
    float s = 0.f;
#pragma unroll 8
    for (int e = 0; e < 128; ++e) s += bf2f(xr[e]) * w[e];
    sUS[(f << 1) + c] = s;  // unary bias omitted: softmax shift-invariant
  }
  __syncthreads();
  if (tid < 2) {
    const int c = tid;
    float mx = -3.0e38f;
    for (int f = 0; f < 128; ++f) mx = fmaxf(mx, sUS[(f << 1) + c]);
    float sum = 0.f;
    for (int f = 0; f < 128; ++f) sum += __expf(sUS[(f << 1) + c] - mx);
    const float inv = 1.0f / sum;
    for (int f = 0; f < 128; ++f)
      sU[(f << 1) + c] = __expf(sUS[(f << 1) + c] - mx) * inv;  // flat = h*128+n
  }

  // ---- Phase 1: residual projection R = X @ Wr^T + br (kept in LDS f32) --
  stage128(sW, g_wr, tid);
  if (tid < 128) sBias[tid] = g_wrb[tid];
  __syncthreads();
  {
    v8f acc[8]; zero_acc(acc, 8);
    gemm_strip<8, 4>(sX, SX_STRIDE, sW, SX_STRIDE, wave, acc, lane);
#pragma unroll
    for (int nt = 0; nt < 8; ++nt)
#pragma unroll
      for (int r = 0; r < 8; ++r) {
        int a = nt * 16 + (lane & 15);
        int f = wave * 16 + r + ((lane & 16) >> 1);
        sR[f * R_STRIDE + a] = acc[nt][r] + sBias[a];
      }
  }
  __syncthreads();

  // ---- Phases 2/3: mean-centered Q and K (biases cancel) ------------------
  qk_project(g_wq, sX, sW, sMuQ, sMeanQ, sCol, tid, lane, wave);
  qk_project(g_wk, sX, sW, sMuK, sMeanK, sCol, tid, lane, wave);

  // ---- Phase 4: V projection (+bias), stored transposed per head ----------
  stage128(sW, g_wv, tid);
  if (tid < 128) sBias[tid] = g_wvb[tid];
  __syncthreads();
  {
    v8f acc[8]; zero_acc(acc, 8);
    gemm_strip<8, 4>(sX, SX_STRIDE, sW, SX_STRIDE, wave, acc, lane);
#pragma unroll
    for (int nt = 0; nt < 8; ++nt)
#pragma unroll
      for (int r = 0; r < 8; ++r) {
        int a = nt * 16 + (lane & 15);
        int f = wave * 16 + r + ((lane & 16) >> 1);
        int h = f >> 6;
        int n = ((f & 63) << 1) + (a >> 6);
        int d = a & 63;
        sVt[(h * 64 + d) * SX_STRIDE + n] = f2bf(acc[nt][r] + sBias[a]);
      }
  }
  __syncthreads();  // last reads of sX/sW were above; sAttn may now alias them

  // ---- Phase 5: pair = softmax(muQ muK^T / 8) + unary -> sAttn (bf16) -----
  for (int si = 0; si < 2; ++si) {
    const int s  = wave + si * 8;      // 16 strips: 2 heads x 8 m-tiles
    const int h  = s >> 3;
    const int mt = s & 7;
    v8f acc[8]; zero_acc(acc, 8);
    gemm_strip<8, 2>(sMuQ + h * 128 * MU_STRIDE, MU_STRIDE,
                     sMuK + h * 128 * MU_STRIDE, MU_STRIDE, mt, acc, lane);
#pragma unroll
    for (int r = 0; r < 8; ++r) {      // softmax over m: row held by 16 lanes
      float x[8]; float mx = -3.0e38f;
#pragma unroll
      for (int nt = 0; nt < 8; ++nt) { x[nt] = acc[nt][r] * 0.125f; mx = fmaxf(mx, x[nt]); }
#pragma unroll
      for (int m1 = 1; m1 < 16; m1 <<= 1) mx = fmaxf(mx, __shfl_xor(mx, m1, 32));
      float sum = 0.f;
#pragma unroll
      for (int nt = 0; nt < 8; ++nt) { x[nt] = __expf(x[nt] - mx); sum += x[nt]; }
#pragma unroll
      for (int m1 = 1; m1 < 16; m1 <<= 1) sum += __shfl_xor(sum, m1, 32);
      const float inv = 1.0f / sum;
      const int n = mt * 16 + r + ((lane & 16) >> 1);
      const float u = sU[h * 128 + n];
      __bf16* arow = sAttn + (h * 128 + n) * SX_STRIDE;
#pragma unroll
      for (int nt = 0; nt < 8; ++nt) arow[nt * 16 + (lane & 15)] = f2bf(x[nt] * inv + u);
    }
  }
  __syncthreads();

  // ---- Phase 6: out = attn @ V + R, remapped back to [f][a] ---------------
  for (int si = 0; si < 2; ++si) {
    const int s  = wave + si * 8;
    const int h  = s >> 3;
    const int mt = s & 7;
    v8f acc[4]; zero_acc(acc, 4);
    gemm_strip<4, 4>(sAttn + h * 128 * SX_STRIDE, SX_STRIDE,
                     sVt   + h * 64  * SX_STRIDE, SX_STRIDE, mt, acc, lane);
#pragma unroll
    for (int nt = 0; nt < 4; ++nt)
#pragma unroll
      for (int r = 0; r < 8; ++r) {
        int n = mt * 16 + r + ((lane & 16) >> 1);
        int d = nt * 16 + (lane & 15);
        int f = (h << 6) + (n >> 1);
        int a = ((n & 1) << 6) + d;
        gout[base + f * 128 + a] = acc[nt][r] + sR[f * R_STRIDE + a];
      }
  }
}

extern "C" void kernel_launch(void* const* d_in, const int* in_sizes, int n_in,
                              void* d_out, int out_size, void* d_ws, size_t ws_size,
                              hipStream_t stream) {
  (void)n_in; (void)out_size; (void)d_ws; (void)ws_size;
  const float* gin  = (const float*)d_in[0];
  const float* g_uw = (const float*)d_in[1];   // unary_w [2,128]
  // d_in[2] unary_b: cancels under softmax
  const float* g_wq = (const float*)d_in[3];   // d_in[4] wq_b cancels (mean-center)
  const float* g_wk = (const float*)d_in[5];   // d_in[6] wk_b cancels
  const float* g_wv = (const float*)d_in[7];
  const float* g_wvb= (const float*)d_in[8];
  const float* g_wr = (const float*)d_in[9];
  const float* g_wrb= (const float*)d_in[10];
  float* gout = (float*)d_out;

  const int blocks = in_sizes[0] >> 14;        // BS = elems / (128*128)

  (void)hipFuncSetAttribute(reinterpret_cast<const void*>(disattn_kernel),
                            hipFuncAttributeMaxDynamicSharedMemorySize,
                            (int)SMEM_BYTES);
  disattn_kernel<<<blocks, 256, SMEM_BYTES, stream>>>(
      gin, g_uw, g_wq, g_wk, g_wv, g_wvb, g_wr, g_wrb, gout);
}
